// DiscretePixelCNN_46042049413290
// MI455X (gfx1250) — matrix operand order repaired and checked by reading, not compiled, not run on hardware
//
#include <hip/hip_runtime.h>
#include <hip/hip_bf16.h>
#include <stdint.h>

// ---------------------------------------------------------------------------
// PixelCNN autoregressive sampler for MI455X (gfx1250, wave32).
// One workgroup (256 threads = 8 waves) per batch row; the entire 256-step
// sampling chain runs inside the kernel. Hidden 3x3 masked convs run as
// V_WMMA_F32_16X16X32_F16 with register-resident *A* (weight) fragments:
// D[out-ch][pixel] = W[out-ch][in-ch] x Act[in-ch][pixel], so each lane's 8
// accumulator elements are 8 contiguous channels of one pixel -> epilogue is
// 4x v_cvt_pk_f16_f32 + ONE ds_store_b128 per tile. All 5 taps' activation
// fragments are prefetched into VGPRs before the back-to-back WMMA chain.
// ---------------------------------------------------------------------------

typedef __attribute__((ext_vector_type(16))) _Float16 v16h;
typedef __attribute__((ext_vector_type(8)))  _Float16 v8h;
typedef __attribute__((ext_vector_type(8)))  float    v8f;

#define HW   16
#define HID  32
#define PADW 18                 // columns -1..16
#define PADH 17                 // rows    -1..15
#define HBUF (PADH * PADW * HID)

__device__ __forceinline__ int hidx(int y, int x) {
    return ((y + 1) * PADW + (x + 1)) * HID;   // halves, 64B-aligned per pixel
}

// mask-A (5x5, type A) valid taps: rows 0,1 full; row 2 cols 0,1
static constexpr int A_KY[12] = {0,0,0,0,0, 1,1,1,1,1, 2,2};
static constexpr int A_KX[12] = {0,1,2,3,4, 0,1,2,3,4, 0,1};
// mask-B (3x3, type B) valid taps: (0,0)(0,1)(0,2)(1,0)(1,1)
static constexpr int B_KY[5]  = {0,0,0,1,1};
static constexpr int B_KX[5]  = {0,1,2,0,1};

struct U2 { uint32_t x, y; };

// Threefry-2x32, 20 rounds (JAX default PRNG core).
__device__ __forceinline__ U2 threefry2x32(uint32_t k0, uint32_t k1,
                                           uint32_t c0, uint32_t c1) {
    static constexpr int R[8] = {13,15,26,6, 17,29,16,24};
    uint32_t ks[3] = {k0, k1, 0x1BD11BDAu ^ k0 ^ k1};
    uint32_t x0 = c0 + ks[0], x1 = c1 + ks[1];
    #pragma unroll
    for (int blk = 1; blk <= 5; ++blk) {
        #pragma unroll
        for (int j = 0; j < 4; ++j) {
            int rr = R[((blk - 1) & 1) * 4 + j];
            x0 += x1;
            x1 = (x1 << rr) | (x1 >> (32 - rr));
            x1 ^= x0;
        }
        x0 += ks[blk % 3];
        x1 += ks[(blk + 1) % 3] + (uint32_t)blk;
    }
    return {x0, x1};
}

__device__ __forceinline__ float gumbel_from_bits(uint32_t bits) {
    float u = (float)(bits >> 8) * (1.0f / 16777216.0f) + 1e-10f;
    return -logf(-logf(u));
}

__global__ __launch_bounds__(256, 1)
void pixelcnn_sample_kernel(
    const float* __restrict__ T,
    const float* __restrict__ w_in,  const float* __restrict__ b_in,
    const float* __restrict__ w_h1,  const float* __restrict__ b_h1,
    const float* __restrict__ w_h2,  const float* __restrict__ b_h2,
    const float* __restrict__ w_h3,  const float* __restrict__ b_h3,
    const float* __restrict__ w_out, const float* __restrict__ b_out,
    float* __restrict__ out)
{
    __shared__ float samp[HW * HW];
    __shared__ __attribute__((aligned(16))) _Float16 h0[HBUF];   // padded f16 acts
    __shared__ __attribute__((aligned(16))) _Float16 h1[HBUF];
    __shared__ float wA[HID][2][12];   // conv1 weights, valid taps only
    __shared__ float bIn[HID];
    __shared__ __attribute__((aligned(16))) float bH[3][HID];  // hidden biases
    __shared__ float wO[2][HID];
    __shared__ float bO[2];
    __shared__ float logitL[2];

    const int tid  = threadIdx.x;
    const int bimg = blockIdx.x;
    const int lane = tid & 31;
    const int wave = tid >> 5;

    const int khalf = lane >> 4;       // K-half select per WMMA ISA layouts
    const int mlan  = lane & 15;       // A: out-ch row / B,D: pixel column
    const int nTile = wave >> 2;       // this wave's fixed 16-channel block
    const int mBase = (wave & 3) * 4;  // this wave's 4 image rows

    const float Tval = T[bimg];

    if (tid == 0) {            // warm L2 for shared weights (global_prefetch_b8)
        __builtin_prefetch(w_h1, 0, 0);
        __builtin_prefetch(w_h2, 0, 0);
        __builtin_prefetch(w_h3, 0, 0);
    }

    // ---- one-time staging: conv1 weights, biases, head weights, halo zero ----
    for (int i = tid; i < HID * 2 * 12; i += 256) {
        int oc = i / 24, rem = i % 24, ic = rem / 12, t = rem % 12;
        wA[oc][ic][t] = w_in[oc * 50 + ic * 25 + A_KY[t] * 5 + A_KX[t]];
    }
    if (tid < HID) {
        bIn[tid]   = b_in[tid];
        bH[0][tid] = b_h1[tid];
        bH[1][tid] = b_h2[tid];
        bH[2][tid] = b_h3[tid];
    }
    if (tid < 2 * HID) wO[tid >> 5][tid & 31] = w_out[tid];
    if (tid < 2)       bO[tid] = b_out[tid];
    if (tid < HW * HW) samp[tid] = 0.0f;
    for (int i = tid; i < HBUF; i += 256) {
        h0[i] = (_Float16)0.0f;   // zero halo persists; interior overwritten
        h1[i] = (_Float16)0.0f;
    }

    // ---- register-resident A (weight) fragments: 3 layers x 5 taps ----------
    // A layout (16-bit, 16x32): lanes 0-15 & 16-31 both hold M = lane&15;
// per-lane K order: e[k] = khalf*16+k (contiguous, matches B).
    v16h Aw[3][5];
    {
        const int outCh = nTile * 16 + mlan;
        #pragma unroll
        for (int l = 0; l < 3; ++l) {
            const float* wl = (l == 0) ? w_h1 : (l == 1) ? w_h2 : w_h3;
            #pragma unroll
            for (int t = 0; t < 5; ++t) {
                union { v16h v; _Float16 e[16]; } u;
                #pragma unroll
                for (int k = 0; k < 16; ++k) {
                    const int inCh = khalf * 16 + k;
                    u.e[k] = (_Float16)wl[outCh * 288 + inCh * 9
                                          + B_KY[t] * 3 + B_KX[t]];
                }
                Aw[l][t] = u.v;
            }
        }
    }
    __syncthreads();

    // ======================= autoregressive pixel loop =======================
    for (int p = 0; p < HW * HW; ++p) {
        // ---- conv1: masked 5x5, 2->32, VALU (tiny K), ReLU -> h0 (f16) ----
        {
            const int y = tid >> 4, x = tid & 15;
            float in0[12], in1[12];
            #pragma unroll
            for (int t = 0; t < 12; ++t) {
                int py = y + A_KY[t] - 2, px = x + A_KX[t] - 2;
                bool ok = (py >= 0) & (px >= 0) & (px < HW);
                in0[t] = ok ? samp[py * HW + px] : 0.0f;
                in1[t] = ok ? Tval : 0.0f;
            }
            #pragma unroll 4
            for (int oc = 0; oc < HID; ++oc) {
                float a = bIn[oc];
                #pragma unroll
                for (int t = 0; t < 12; ++t)
                    a = fmaf(in0[t], wA[oc][0][t], fmaf(in1[t], wA[oc][1][t], a));
                h0[hidx(y, x) + oc] = (_Float16)fmaxf(a, 0.0f);
            }
        }
        __syncthreads();

        // ---- three masked 3x3 hidden layers via WMMA (weights in VGPRs) ----
        #pragma unroll
        for (int l = 0; l < 3; ++l) {
            _Float16* hin  = (l == 1) ? h1 : h0;   // h0->h1->h0->h1
            _Float16* hout = (l == 1) ? h0 : h1;
            const int chBase = nTile * 16 + khalf * 8;   // D channel base/lane
            #pragma unroll
            for (int ti = 0; ti < 4; ++ti) {
                const int m = mBase + ti;          // image row of this tile
                // Prefetch all 5 taps' activation (B) fragments: 10 in-flight
                // ds_load_b128, then 5 back-to-back WMMAs.
                // B layout (16-bit, 32x16): lane N = pixel (lane&15);
                // per-lane K = khalf*16 + 0..15 -> 16 contiguous halves.
                v16h Ba[5];
                #pragma unroll
                for (int t = 0; t < 5; ++t) {
                    const int dy = B_KY[t] - 1, dx = B_KX[t] - 1;
                    const _Float16* bp =
                        hin + hidx(m + dy, mlan + dx) + khalf * 16;
                    union { v16h v; v8h h[2]; } U;
                    U.h[0] = *(const v8h*)(bp);
                    U.h[1] = *(const v8h*)(bp + 8);
                    Ba[t] = U.v;
                }
                v8f acc = {};
                #pragma unroll
                for (int t = 0; t < 5; ++t)
                    acc = __builtin_amdgcn_wmma_f32_16x16x32_f16(
                              false, Aw[l][t], false, Ba[t],
                              (short)0, acc, false, false);
                // bias + ReLU + pack; lane's 8 D elements = 8 contiguous
                // channels of pixel (m, mlan) -> single ds_store_b128.
                union { v8h v; _Float16 e[8]; } o;
                #pragma unroll
                for (int j = 0; j < 8; ++j)
                    o.e[j] = (_Float16)fmaxf(acc[j] + bH[l][chBase + j], 0.0f);
                *(v8h*)(hout + hidx(m, mlan) + chBase) = o.v;
            }
            __syncthreads();
        }
        // final hidden activations are in h1

        // ---- 1x1 head: only the current pixel's 2 logits are needed ----
        if (tid < 2) {
            const int py = p >> 4, px = p & 15;
            float a = bO[tid];
            const _Float16* hp = h1 + hidx(py, px);
            #pragma unroll
            for (int c = 0; c < HID; ++c) a += (float)hp[c] * wO[tid][c];
            logitL[tid] = a;
        }
        __syncthreads();

        // ---- categorical sample (Threefry fold_in + Gumbel argmax) ----
        if (tid == 0) {
            float val;
            if (p == 0) {
                val = 1.0f;                                  // fix_first
            } else {
                U2 fk = threefry2x32(0u, 42u, 0u, (uint32_t)p);          // fold_in
                U2 r  = threefry2x32(fk.x, fk.y, 0u, (uint32_t)(bimg * 2));
                float g0 = gumbel_from_bits(r.x);
                float g1 = gumbel_from_bits(r.y);
                val = (logitL[1] + g1 > logitL[0] + g0) ? 1.0f : 0.0f;
            }
            samp[p] = val;
        }
        __syncthreads();
    }

    if (tid < HW * HW) out[bimg * HW * HW + tid] = samp[tid];
}

extern "C" void kernel_launch(void* const* d_in, const int* in_sizes, int n_in,
                              void* d_out, int out_size, void* d_ws, size_t ws_size,
                              hipStream_t stream) {
    (void)n_in; (void)out_size; (void)d_ws; (void)ws_size;
    const float* T     = (const float*)d_in[0];
    const float* w_in  = (const float*)d_in[1];
    const float* b_in  = (const float*)d_in[2];
    const float* w_h1  = (const float*)d_in[3];
    const float* b_h1  = (const float*)d_in[4];
    const float* w_h2  = (const float*)d_in[5];
    const float* b_h2  = (const float*)d_in[6];
    const float* w_h3  = (const float*)d_in[7];
    const float* b_h3  = (const float*)d_in[8];
    const float* w_out = (const float*)d_in[9];
    const float* b_out = (const float*)d_in[10];

    const int B = in_sizes[0];   // 64 batch rows -> 64 independent workgroups
    hipLaunchKernelGGL(pixelcnn_sample_kernel, dim3(B), dim3(256), 0, stream,
                       T, w_in, b_in, w_h1, b_h1, w_h2, b_h2, w_h3, b_h3,
                       w_out, b_out, (float*)d_out);
}
